// ClusterAssignment_15874199126460
// MI455X (gfx1250) — compile-verified, matrix operand order checked by reading, not run on hardware
//
#include <hip/hip_runtime.h>
#include <hip/hip_bf16.h>

// ---------------------------------------------------------------------------
// Soft cluster assignment (Student-t, alpha=1) for MI455X / gfx1250.
//   cross = batch @ centers^T  via V_WMMA_F32_16X16X32_BF16, bf16x3 split
//   out   = (1/(1+max(|x|^2 - 2 cross + |c|^2, 0))) row-normalized
// N=65536, K=1024, D=512 (fp32 in/out).
// ---------------------------------------------------------------------------

typedef __bf16 bf16_t;
typedef __attribute__((ext_vector_type(16))) __bf16 v16bf;
typedef __attribute__((ext_vector_type(8)))  __bf16 v8bf;
typedef __attribute__((ext_vector_type(8)))  float  v8f;

#define WMMA_BF16(A, B, C) \
  __builtin_amdgcn_wmma_f32_16x16x32_bf16(false, (A), false, (B), (short)0, (C), false, false)

static constexpr int NROWS  = 65536;
static constexpr int KCENT  = 1024;
static constexpr int DDIM   = 512;
static constexpr int MT     = 16;            // rows per workgroup
static constexpr int AST    = 520;           // LDS row stride in bf16 (1040B: 4-bank rotate)
static constexpr int CHUNKS = DDIM / 32;     // 16 k-chunks
static constexpr int TILES  = KCENT / 16;    // 64 column tiles
static constexpr int TPW    = TILES / 8;     // 8 tiles per wave (8 waves/block)

__device__ __forceinline__ void split_bf16(float x, __bf16& hi, __bf16& lo) {
  hi = (__bf16)x;                 // RNE f32 -> bf16
  lo = (__bf16)(x - (float)hi);   // residual
}

// ---------------------------------------------------------------------------
// Prep 1: c_sq[k] = sum_d centers[k][d]^2, exact fp32. One wave per row.
// ---------------------------------------------------------------------------
__global__ __launch_bounds__(256) void prep_csq(const float* __restrict__ centers,
                                                float* __restrict__ csq) {
  const int w    = (blockIdx.x * blockDim.x + threadIdx.x) >> 5;  // row id
  const int lane = threadIdx.x & 31;
  if (w < KCENT) {
    const float* row = centers + (size_t)w * DDIM;
    float s = 0.f;
    #pragma unroll 4
    for (int i = lane; i < DDIM; i += 32) { float x = row[i]; s += x * x; }
    #pragma unroll
    for (int off = 16; off > 0; off >>= 1) s += __shfl_xor(s, off, 32);
    if (lane == 0) csq[w] = s;
  }
}

// ---------------------------------------------------------------------------
// Prep 2: repack centers into WMMA B-fragment order, split into hi/lo bf16.
// Fragment slot (chunk c, tile t, lane l) holds 16 bf16:
//   s in 0..7  : k = c*32 + 8*(l>>4) + s       (contiguous, 16B)
//   s in 8..15 : k = c*32 + 8*(l>>4) + 16 + s-8 (contiguous, 16B)
//   column (center row) = t*16 + (l&15)
// Flat address: ((c*TILES + t)*32 + l)*16 elements.
// ---------------------------------------------------------------------------
__global__ __launch_bounds__(256) void prep_pack(const float* __restrict__ centers,
                                                 bf16_t* __restrict__ Bhi,
                                                 bf16_t* __restrict__ Blo) {
  const int T = blockIdx.x * 256 + threadIdx.x;        // 0 .. 16*64*32-1
  const int c = T >> 11;                               // / (TILES*32)
  const int t = (T >> 5) & (TILES - 1);
  const int l = T & 31;
  const int half = l >> 4, m = l & 15;
  const int row = t * 16 + m;
  const float* src = centers + (size_t)row * DDIM + c * 32 + 8 * half;
  bf16_t* dh = Bhi + (size_t)T * 16;
  bf16_t* dl = Blo + (size_t)T * 16;
  #pragma unroll
  for (int u = 0; u < 2; ++u) {
    #pragma unroll
    for (int j = 0; j < 8; ++j) {
      __bf16 h, lo;
      split_bf16(src[u * 16 + j], h, lo);
      dh[u * 8 + j] = h;
      dl[u * 8 + j] = lo;
    }
  }
}

// ---------------------------------------------------------------------------
// Main: each block = 16 rows x 1024 cols. 8 waves, wave w owns cols
// [w*128, w*128+128) as 8 accumulator tiles. bf16x3 WMMA over 16 k-chunks.
// ---------------------------------------------------------------------------
__global__ __launch_bounds__(256) void cluster_assign(const float* __restrict__ batch,
                                                      const bf16_t* __restrict__ Bhi,
                                                      const bf16_t* __restrict__ Blo,
                                                      const float* __restrict__ csq_g,
                                                      float* __restrict__ out) {
  __shared__ __align__(16) __bf16 Ahi[MT * AST];
  __shared__ __align__(16) __bf16 Alo[MT * AST];
  __shared__ float csq_s[KCENT];
  __shared__ float xsq_s[MT];
  __shared__ float rowsum_s[MT];

  const int tid  = threadIdx.x;
  const int lane = tid & 31;
  const int wave = tid >> 5;
  const int half = lane >> 4;
  const int m    = lane & 15;
  const size_t rowBase = (size_t)blockIdx.x * MT;

  if (tid < MT) { xsq_s[tid] = 0.f; rowsum_s[tid] = 0.f; }
  __syncthreads();

  // ---- stage A rows into LDS as hi/lo bf16; accumulate |x|^2 in fp32 ----
  const float* bsrc = batch + rowBase * DDIM;
  #pragma unroll 4
  for (int i = 0; i < (MT * DDIM) / 256; ++i) {
    const int e = i * 256 + tid;
    const int r = e >> 9;             // /DDIM
    const int col = e & (DDIM - 1);
    const float x = __builtin_nontemporal_load(&bsrc[e]);
    __bf16 h, lo;
    split_bf16(x, h, lo);
    Ahi[r * AST + col] = h;
    Alo[r * AST + col] = lo;
    atomicAdd(&xsq_s[r], x * x);      // ds_add_f32
  }
  // ---- stage c_sq (L2-hot, 4KB) ----
  #pragma unroll
  for (int i = 0; i < KCENT / 256; ++i) csq_s[i * 256 + tid] = csq_g[i * 256 + tid];
  __syncthreads();

  v8f acc[TPW] = {};
  const int gt0 = wave * TPW;

  for (int c = 0; c < CHUNKS; ++c) {
    // A fragments: two 16B LDS reads each (bank-rotated rows, conflict-light)
    const int abase = m * AST + c * 32 + 8 * half;
    v8bf ah0 = *(const v8bf*)&Ahi[abase];
    v8bf ah1 = *(const v8bf*)&Ahi[abase + 16];
    v8bf al0 = *(const v8bf*)&Alo[abase];
    v8bf al1 = *(const v8bf*)&Alo[abase + 16];
    v16bf ahi = __builtin_shufflevector(ah0, ah1, 0,1,2,3,4,5,6,7,8,9,10,11,12,13,14,15);
    v16bf alo = __builtin_shufflevector(al0, al1, 0,1,2,3,4,5,6,7,8,9,10,11,12,13,14,15);

    const size_t slotBase = (size_t)c * (TILES * 32) + (size_t)gt0 * 32 + lane;

    // B fragments: 32B/lane coalesced, register double-buffered across tiles
    v16bf bh = *(const v16bf*)(Bhi + slotBase * 16);
    v16bf bl = *(const v16bf*)(Blo + slotBase * 16);
    #pragma unroll
    for (int t = 0; t < TPW; ++t) {
      v16bf bhN = bh, blN = bl;
      if (t + 1 < TPW) {
        bhN = *(const v16bf*)(Bhi + (slotBase + (size_t)(t + 1) * 32) * 16);
        blN = *(const v16bf*)(Blo + (slotBase + (size_t)(t + 1) * 32) * 16);
      }
      if (c + 1 < CHUNKS) {  // pull next chunk's stream toward the WGP
        __builtin_prefetch(Bhi + (slotBase + (size_t)(TILES * 32) + (size_t)t * 32) * 16, 0, 1);
        __builtin_prefetch(Blo + (slotBase + (size_t)(TILES * 32) + (size_t)t * 32) * 16, 0, 1);
      }
      // cross ~= hi*hi + hi*lo + lo*hi  (drop lo*lo): ~fp32 accuracy
      acc[t] = WMMA_BF16(ahi, bh, acc[t]);
      acc[t] = WMMA_BF16(ahi, bl, acc[t]);
      acc[t] = WMMA_BF16(alo, bh, acc[t]);
      bh = bhN; bl = blN;
    }
  }

  // ---- epilogue: Student-t numerator + block-local row normalization ----
  float xs[8];
  #pragma unroll
  for (int v = 0; v < 8; ++v) xs[v] = xsq_s[v + 8 * half];  // row = v + 8*half

  float psum[8] = {0.f,0.f,0.f,0.f,0.f,0.f,0.f,0.f};
  #pragma unroll
  for (int t = 0; t < TPW; ++t) {
    const float cs = csq_s[(gt0 + t) * 16 + m];
    #pragma unroll
    for (int v = 0; v < 8; ++v) {
      float ns = xs[v] - 2.f * acc[t][v] + cs;
      ns = ns < 0.f ? 0.f : ns;                 // jnp.maximum(ns, 0)
      const float num = 1.f / (1.f + ns);       // alpha=1 => power==1, no pow
      acc[t][v] = num;
      psum[v] += num;
    }
  }
  #pragma unroll
  for (int v = 0; v < 8; ++v) atomicAdd(&rowsum_s[v + 8 * half], psum[v]);
  __syncthreads();

  float inv[8];
  #pragma unroll
  for (int v = 0; v < 8; ++v) inv[v] = 1.f / rowsum_s[v + 8 * half];

  #pragma unroll
  for (int t = 0; t < TPW; ++t) {
    const size_t colBase = (size_t)(gt0 + t) * 16 + m;
    #pragma unroll
    for (int v = 0; v < 8; ++v) {
      const size_t r = rowBase + v + 8 * half;
      __builtin_nontemporal_store(acc[t][v] * inv[v], &out[r * KCENT + colBase]);
    }
  }
}

// ---------------------------------------------------------------------------
// Launch: prep (csq + fragment repack into d_ws) then fused GEMM/assignment.
// Workspace: [0,1MB) Bhi, [1MB,2MB) Blo, [2MB,+4KB) c_sq.
// ---------------------------------------------------------------------------
extern "C" void kernel_launch(void* const* d_in, const int* in_sizes, int n_in,
                              void* d_out, int out_size, void* d_ws, size_t ws_size,
                              hipStream_t stream) {
  const float* batch   = (const float*)d_in[0];   // [N, D] f32
  const float* centers = (const float*)d_in[1];   // [K, D] f32
  float* out = (float*)d_out;                     // [N, K] f32

  char* ws = (char*)d_ws;
  bf16_t* Bhi = (bf16_t*)ws;                        // 1 MB
  bf16_t* Blo = (bf16_t*)(ws + (1u << 20));         // 1 MB
  float*  csq = (float*)(ws + (2u << 20));          // 4 KB

  prep_csq <<<(KCENT * 32) / 256, 256, 0, stream>>>(centers, csq);
  prep_pack<<<(CHUNKS * TILES * 32) / 256, 256, 0, stream>>>(centers, Bhi, Blo);
  cluster_assign<<<NROWS / MT, 256, 0, stream>>>(batch, Bhi, Blo, csq, out);
}
